// Curve_64690797412573
// MI455X (gfx1250) — matrix operand order repaired and checked by reading, not compiled, not run on hardware
//
#include <hip/hip_runtime.h>
#include <hip/hip_bf16.h>

// CDNA5 / gfx1250. wave32.
typedef __attribute__((ext_vector_type(16))) _Float16 v16h;
typedef __attribute__((ext_vector_type(8)))  float    v8f;

#define NUM_BINS 256
#define PLANES   48          // 16 batches * 3 channels
#define HW       (1024*1024) // pixels per plane

// ---------------------------------------------------------------------------
// Kernel 1: curve = minmax_norm( sigmoid(x) @ triu(ones(256,256)) )
// One workgroup, 256 threads (8 waves). 48 output 16x16 tiles, 6 per wave.
// cumsum done as WMMA matmul against an in-register upper-triangular 0/1
// B-tile. f32 accuracy recovered via hi/lo f16 splitting of the A operand
// (two chained v_wmma per k-step into the same f32 accumulator).
// Rows of the LUT: row = channel*16 + batch.
// ---------------------------------------------------------------------------
__global__ __launch_bounds__(256) void curve_wmma_kernel(
    const float* __restrict__ rr, const float* __restrict__ gr,
    const float* __restrict__ br, float* __restrict__ curve /* [48*256] */) {

  __shared__ float curveS[PLANES * NUM_BINS];   // 48 KB of the 320 KB WGP LDS
  __shared__ float mnS[3], scS[3];

  const int tid    = threadIdx.x;
  const int lane   = tid & 31;
  const int wave   = tid >> 5;
  const int m      = lane & 15;    // row within tile (A) / column N (B,C,D)
  const int laneHi = lane >> 4;    // lane-half selector

  for (int tile = wave; tile < 48; tile += 8) {
    const int mTile = tile >> 4;   // == channel (rows 16c..16c+15)
    const int nTile = tile & 15;   // bin-column tile
    const float* src = (mTile == 0 ? rr : (mTile == 1 ? gr : br)) + m * NUM_BINS;

    v8f acc = {0.f, 0.f, 0.f, 0.f, 0.f, 0.f, 0.f, 0.f};

    for (int kt = 0; kt < 8; ++kt) {          // K = 256 in steps of 32
      const int kBase = kt * 32;
      v16h ahi, alo, bt;
#pragma unroll
      for (int h = 0; h < 16; ++h) {
        // A 16x32 f16 layout: slot h -> K
        const int kA = (h < 8) ? (laneHi * 8 + h) : (16 + laneHi * 8 + (h - 8));
        const float x = src[kBase + kA];
        const float s = 1.0f / (1.0f + __expf(-x));   // sigmoid
        const _Float16 hi = (_Float16)s;
        ahi[h] = hi;
        alo[h] = (_Float16)(s - (float)hi);           // f16 rounding residual
        // B 32x16 f16 layout: slot h -> K = kA (same as A), column N = lane&15
        const int kB = kBase + kA;
        const int ng = nTile * 16 + m;
        bt[h] = (kB <= ng) ? (_Float16)1.0f : (_Float16)0.0f;  // triu(ones)
      }
      // D = A*B + C; (neg_a, A, neg_b, B, c_mod, C, reuse_a, reuse_b)
      acc = __builtin_amdgcn_wmma_f32_16x16x32_f16(false, ahi, false, bt,
                                                   (short)0, acc, false, false);
      acc = __builtin_amdgcn_wmma_f32_16x16x32_f16(false, alo, false, bt,
                                                   (short)0, acc, false, false);
    }
    // C/D 16x16 f32 layout: VGPR v -> M = v + 8*laneHi, N = lane&15
#pragma unroll
    for (int v = 0; v < 8; ++v) {
      const int row = mTile * 16 + v + laneHi * 8;
      const int col = nTile * 16 + m;
      curveS[row * NUM_BINS + col] = acc[v];
    }
  }
  __syncthreads();

  // cumsum rows are strictly increasing -> channel min lives in bin 0,
  // channel max in bin 255; reduce the 16 batch rows per channel.
  if (tid < 3) {
    float mn = curveS[(tid * 16) * NUM_BINS + 0];
    float mx = curveS[(tid * 16) * NUM_BINS + 255];
    for (int b = 1; b < 16; ++b) {
      mn = fminf(mn, curveS[(tid * 16 + b) * NUM_BINS + 0]);
      mx = fmaxf(mx, curveS[(tid * 16 + b) * NUM_BINS + 255]);
    }
    mnS[tid] = mn;
    scS[tid] = 1.0f / (mx - mn);
  }
  __syncthreads();

  for (int e = tid; e < PLANES * NUM_BINS; e += 256) {
    const int c = (e >> 8) >> 4;
    curve[e] = (curveS[e] - mnS[c]) * scS[c];
  }
}

// ---------------------------------------------------------------------------
// Kernel 2: streaming LUT gather. HBM-bound: 403 MB traffic -> ~17 us ceiling
// at 23.3 TB/s. One block per 2048-pixel chunk of one (b,c) plane; the plane's
// 1 KB LUT is staged in LDS so the random gather never touches global memory.
// float4 loads/stores keep the global side at 16B/lane coalesced B128 ops.
// ---------------------------------------------------------------------------
__global__ __launch_bounds__(256) void lut_gather_kernel(
    const float* __restrict__ img, const float* __restrict__ curve,
    float* __restrict__ out) {

  __shared__ float lut[NUM_BINS];
  const int tid   = threadIdx.x;
  const int plane = blockIdx.y;        // b*3 + c  (img memory order)
  const int b     = plane / 3;
  const int c     = plane % 3;
  lut[tid] = curve[(c * 16 + b) * NUM_BINS + tid];
  __syncthreads();

  const size_t base = (size_t)plane * HW + (size_t)blockIdx.x * 2048;
  const float4* __restrict__ ip = (const float4*)(img + base);
  float4*       __restrict__ op = (float4*)(out + base);

#pragma unroll
  for (int i = 0; i < 2; ++i) {
    const int vi = i * 256 + tid;      // float4 index within the chunk
    const float4 v = ip[vi];
    // clip(x*255, 0, 255) then truncate-toward-zero (v_cvt_i32_f32 == int cast)
    const int i0 = (int)fminf(fmaxf(v.x * 255.0f, 0.0f), 255.0f);
    const int i1 = (int)fminf(fmaxf(v.y * 255.0f, 0.0f), 255.0f);
    const int i2 = (int)fminf(fmaxf(v.z * 255.0f, 0.0f), 255.0f);
    const int i3 = (int)fminf(fmaxf(v.w * 255.0f, 0.0f), 255.0f);
    float4 r;
    r.x = lut[i0]; r.y = lut[i1]; r.z = lut[i2]; r.w = lut[i3];
    op[vi] = r;
  }
}

// ---------------------------------------------------------------------------
extern "C" void kernel_launch(void* const* d_in, const int* in_sizes, int n_in,
                              void* d_out, int out_size, void* d_ws, size_t ws_size,
                              hipStream_t stream) {
  const float* img = (const float*)d_in[0];  // [16,3,1024,1024]
  const float* rr  = (const float*)d_in[1];  // [16,256]
  const float* gr  = (const float*)d_in[2];
  const float* br  = (const float*)d_in[3];
  float* out   = (float*)d_out;
  float* curve = (float*)d_ws;               // 48*256 f32 = 49152 B scratch

  hipLaunchKernelGGL(curve_wmma_kernel, dim3(1), dim3(256), 0, stream,
                     rr, gr, br, curve);

  // 48 planes * (1M pixels / 2048 per block) = 48 x 512 blocks
  hipLaunchKernelGGL(lut_gather_kernel, dim3(512, 48), dim3(256), 0, stream,
                     img, curve, out);
}
